// ContrastiveLoss_14654428414813
// MI455X (gfx1250) — matrix operand order repaired and checked by reading, not compiled, not run on hardware
//
#include <hip/hip_runtime.h>
#include <math.h>

// ---------------------------------------------------------------------------
// Contrastive loss for MI455X (gfx1250, wave32, WMMA).
//   features [1024,512,8,8] f32  -> spatial mean -> L2 normalize (fp16)
//   sim = fn @ fn^T via V_WMMA_F32_16X16X32_F16, fused exp/mask/row-sum
//   loss = mean(log(denom) - log(pos))
// HBM roofline: dominated by the 134MB feature read (~5.8us @ 23.3 TB/s).
// Gram operand fn is 1MB fp16 -> fully L2 resident; WMMA work is ~1 GFLOP.
// ---------------------------------------------------------------------------

typedef _Float16 v16h __attribute__((ext_vector_type(16)));
typedef _Float16 v8h  __attribute__((ext_vector_type(8)));
typedef _Float16 h2   __attribute__((ext_vector_type(2)));
typedef float    v8f  __attribute__((ext_vector_type(8)));

#define BATCH 1024
#define CH    512
#define HW    64           // 8*8 spatial
#define TEMP_INV 2.0f      // 1 / 0.5
#define EPS_NORM 1e-8f

// ---------------------------------------------------------------------------
// Kernel 1: spatial mean + L2 normalize. One block per batch row b.
// Coalesced: each iteration the 256 threads read 4KB contiguous; every
// aligned 16-lane group reduces exactly one channel (64 floats) via shfl.
// Writes fp16 fn (row-major, WMMA A operand) and fnT (transposed; not read by the Gram kernel).
// ---------------------------------------------------------------------------
__global__ __launch_bounds__(256) void mean_norm_kernel(
    const float* __restrict__ feats,
    _Float16* __restrict__ fn,
    _Float16* __restrict__ fnT) {
  __shared__ float smem_mean[CH];
  __shared__ float smem_red[256];
  const int b = blockIdx.x;
  const int t = threadIdx.x;
  const float4* row = (const float4*)(feats + (size_t)b * (CH * HW));

  // phase 1: per-channel means (16 float4 per channel, 16 lanes per channel)
  #pragma unroll 4
  for (int k = 0; k < 32; ++k) {
    const int idx = k * 256 + t;          // float4 index within the row
    float4 v = row[idx];
    float s = v.x + v.y + v.z + v.w;
    s += __shfl_xor(s, 1);
    s += __shfl_xor(s, 2);
    s += __shfl_xor(s, 4);
    s += __shfl_xor(s, 8);                // sum over the 16-lane group
    if ((t & 15) == 0) smem_mean[idx >> 4] = s * (1.0f / HW);
  }
  __syncthreads();

  // phase 2: ||f_b||^2 over 512 channels (2 channels per thread)
  const float m0 = smem_mean[2 * t + 0];
  const float m1 = smem_mean[2 * t + 1];
  smem_red[t] = m0 * m0 + m1 * m1;
  __syncthreads();
  for (int off = 128; off > 0; off >>= 1) {
    if (t < off) smem_red[t] += smem_red[t + off];
    __syncthreads();
  }
  const float scale = 1.0f / fmaxf(sqrtf(smem_red[0]), EPS_NORM);

  const _Float16 h0 = (_Float16)(m0 * scale);
  const _Float16 h1 = (_Float16)(m1 * scale);
  h2 hv; hv.x = h0; hv.y = h1;
  *(h2*)(fn + (size_t)b * CH + 2 * t) = hv;        // row-major (A operand)
  fnT[(size_t)(2 * t + 0) * BATCH + b] = h0;       // transposed (unused by gram_loss_kernel)
  fnT[(size_t)(2 * t + 1) * BATCH + b] = h1;
}

// ---------------------------------------------------------------------------
// Kernel 2: fused Gram + exp + masked row-sums via WMMA f32_16x16x32_f16.
// 1024 waves; wave gw -> i-tile ti = gw>>4 (16 rows), j-group tjg = gw&15
// (4 consecutive 16-col tiles -> A reused 4x). K = 512 -> 16 WMMA steps.
//
// Operand layouts per CDNA5 ISA 7.12.2 (wave32):
//   A 16x32 f16 : lane L holds row M=L&15; halfs {0..7}->K=h*8+q,
//                 {8..15}->K=16+h*8+q, h=L>>4   (two 16B loads from fn)
//   B 32x16 f16 : lane L holds N=L&15; same K order as A (two 16B loads from fn)
//   C/D 16x16   : lane L col N=L&15; VGPR v row M=v+8*(L>>4)
// ---------------------------------------------------------------------------
__global__ __launch_bounds__(256) void gram_loss_kernel(
    const _Float16* __restrict__ fn,
    const _Float16* __restrict__ fnT,
    const int* __restrict__ labels,
    float* __restrict__ pos,
    float* __restrict__ denom) {
  const int tid   = threadIdx.x;
  const int lane  = tid & 31;
  const int gw    = blockIdx.x * 8 + (tid >> 5);
  const int iBase = (gw >> 4) * 16;
  const int jBase = (gw & 15) * 64;
  const int h     = lane >> 4;
  const int m     = lane & 15;

  v8f acc[4];
  #pragma unroll
  for (int jj = 0; jj < 4; ++jj) acc[jj] = 0.0f;

  const _Float16* arow  = fn  + (size_t)(iBase + m) * CH + h * 8;
  const _Float16* bbase = fn + (size_t)(jBase + m) * CH + h * 8;

  #pragma unroll 2
  for (int kk = 0; kk < CH; kk += 32) {
    v8h alo = *(const v8h*)(arow + kk);
    v8h ahi = *(const v8h*)(arow + kk + 16);
    v16h a;
    #pragma unroll
    for (int q = 0; q < 8; ++q) { a[q] = alo[q]; a[8 + q] = ahi[q]; }

    const _Float16* bp = bbase + kk;
    #pragma unroll
    for (int jj = 0; jj < 4; ++jj) {
      const _Float16* br = bp + (size_t)jj * 16 * CH;
      v8h blo = *(const v8h*)(br);
      v8h bhi = *(const v8h*)(br + 16);
      v16h bmat;
      #pragma unroll
      for (int q = 0; q < 8; ++q) { bmat[q] = blo[q]; bmat[8 + q] = bhi[q]; }
      acc[jj] = __builtin_amdgcn_wmma_f32_16x16x32_f16(
          false, a, false, bmat, (short)0, acc[jj], false, false);
    }
  }

  // epilogue: e = exp(2*sim); pos adds only label-matching columns.
  int li[8];
  #pragma unroll
  for (int v = 0; v < 8; ++v) li[v] = labels[iBase + 8 * h + v];

  float dsum[8], psum[8];
  #pragma unroll
  for (int v = 0; v < 8; ++v) { dsum[v] = 0.0f; psum[v] = 0.0f; }

  #pragma unroll
  for (int jj = 0; jj < 4; ++jj) {
    const int lj = labels[jBase + jj * 16 + m];   // this lane's column label
    #pragma unroll
    for (int v = 0; v < 8; ++v) {
      float e = __expf(acc[jj][v] * TEMP_INV);
      float p = (li[v] == lj) ? e : 0.0f;
      e += __shfl_xor(e, 1); p += __shfl_xor(p, 1);
      e += __shfl_xor(e, 2); p += __shfl_xor(p, 2);
      e += __shfl_xor(e, 4); p += __shfl_xor(p, 4);
      e += __shfl_xor(e, 8); p += __shfl_xor(p, 8);   // sum over 16 cols
      if (m == 0) { dsum[v] += e; psum[v] += p; }
    }
  }
  if (m == 0) {
    #pragma unroll
    for (int v = 0; v < 8; ++v) {
      const int i = iBase + 8 * h + v;
      atomicAdd(&denom[i], dsum[v]);
      atomicAdd(&pos[i],   psum[v]);
    }
  }
}

// ---------------------------------------------------------------------------
// Kernel 3: loss = mean(log(denom) - log(pos))
// ---------------------------------------------------------------------------
__global__ __launch_bounds__(256) void loss_reduce_kernel(
    const float* __restrict__ pos,
    const float* __restrict__ denom,
    float* __restrict__ out) {
  __shared__ float red[256];
  const int t = threadIdx.x;
  float a = 0.0f;
  for (int i = t; i < BATCH; i += 256)
    a += __logf(denom[i]) - __logf(pos[i]);
  red[t] = a;
  __syncthreads();
  for (int off = 128; off > 0; off >>= 1) {
    if (t < off) red[t] += red[t + off];
    __syncthreads();
  }
  if (t == 0) out[0] = red[0] * (1.0f / BATCH);
}

extern "C" void kernel_launch(void* const* d_in, const int* in_sizes, int n_in,
                              void* d_out, int out_size, void* d_ws, size_t ws_size,
                              hipStream_t stream) {
  (void)in_sizes; (void)n_in; (void)out_size; (void)ws_size;
  const float* feats  = (const float*)d_in[0];
  const int*   labels = (const int*)d_in[1];   // jax default: int32

  char* ws = (char*)d_ws;
  float*    pos   = (float*)ws;                                 // 1024 f32
  float*    denom = pos + BATCH;                                // 1024 f32
  _Float16* fn    = (_Float16*)(ws + 8192);                     // 1 MB
  _Float16* fnT   = (_Float16*)(ws + 8192 +
                                (size_t)BATCH * CH * sizeof(_Float16)); // 1 MB

  hipMemsetAsync(ws, 0, 2 * BATCH * sizeof(float), stream);     // zero accums
  mean_norm_kernel<<<BATCH, 256, 0, stream>>>(feats, fn, fnT);
  gram_loss_kernel<<<128, 256, 0, stream>>>(fn, fnT, labels, pos, denom);
  loss_reduce_kernel<<<1, 256, 0, stream>>>(pos, denom, (float*)d_out);
}
